// cls_model_3264175145575
// MI455X (gfx1250) — matrix operand order repaired and checked by reading, not compiled, not run on hardware
//
#include <hip/hip_runtime.h>
#include <hip/hip_bf16.h>
#include <math.h>

// ---------------------------------------------------------------------------
// Types for CDNA5 WMMA (wave32)
// ---------------------------------------------------------------------------
typedef __attribute__((ext_vector_type(16))) __bf16 bf16x16;
typedef __attribute__((ext_vector_type(8)))  __bf16 bf16x8;
typedef __attribute__((ext_vector_type(8)))  float  f32x8;

#define NN   4096      // nodes
#define CP   1024      // padded channel count (1000 -> 1024)
#define CV   1000      // valid channels
#define XH2K 6144      // 6 * 1024 (cheb-2 concat K, padded)
#define NR_  20
#define S_   205

// ---------------------------------------------------------------------------
// WMMA fragment loaders from global (bf16, 16x16x32). A: [M,K] row-major.
// Bt: [N,K] row-major. Layouts per CDNA5 ISA 7.12.2.
// ---------------------------------------------------------------------------
__device__ __forceinline__ bf16x16 frag_a(const __bf16* A, long lda, int m0,
                                          int k0, int lane) {
  int m = lane & 15, h = lane >> 4;
  const __bf16* p = A + (long)(m0 + m) * lda + k0 + 8 * h;
  bf16x8 lo = *(const bf16x8*)p;          // k = 8h .. 8h+7
  bf16x8 hi = *(const bf16x8*)(p + 16);   // k = 16+8h .. 16+8h+7
  bf16x16 r;
#pragma unroll
  for (int i = 0; i < 8; ++i) { r[i] = lo[i]; r[i + 8] = hi[i]; }
  return r;
}

__device__ __forceinline__ bf16x16 frag_b(const __bf16* Bt, long ldb, int n0,
                                          int k0, int lane) {
  int n = lane & 15, h = lane >> 4;
  const __bf16* p = Bt + (long)(n0 + n) * ldb + k0 + 8 * h; // k = 8h..8h+7, 16+8h..
  bf16x8 lo = *(const bf16x8*)p;
  bf16x8 hi = *(const bf16x8*)(p + 16);
  bf16x16 r;
#pragma unroll
  for (int i = 0; i < 8; ++i) { r[i] = lo[i]; r[i + 8] = hi[i]; }
  return r;
}

// ---------------------------------------------------------------------------
// Fallback direct-global GEMM (kept compiled; not used on heavy paths).
// ---------------------------------------------------------------------------
__global__ __launch_bounds__(256) void gemm_abt_bf16(
    const __bf16* __restrict__ A, const __bf16* __restrict__ Bt,
    float* __restrict__ C, int M, int N, int K, int lda, int ldb, int ldc) {
  (void)M; (void)N;
  int lane = threadIdx.x & 31;
  int wave = threadIdx.x >> 5;
  int m0 = blockIdx.y * 128 + (wave & 3) * 32;
  int n0 = blockIdx.x * 64 + (wave >> 2) * 32;
  f32x8 acc00 = {}, acc01 = {}, acc10 = {}, acc11 = {};
  for (int k0 = 0; k0 < K; k0 += 32) {
    bf16x16 a0 = frag_a(A, lda, m0, k0, lane);
    bf16x16 a1 = frag_a(A, lda, m0 + 16, k0, lane);
    bf16x16 b0 = frag_b(Bt, ldb, n0, k0, lane);
    bf16x16 b1 = frag_b(Bt, ldb, n0 + 16, k0, lane);
    acc00 = __builtin_amdgcn_wmma_f32_16x16x32_bf16(false, a0, false, b0,
                                                    (short)0, acc00, false, false);
    acc01 = __builtin_amdgcn_wmma_f32_16x16x32_bf16(false, a0, false, b1,
                                                    (short)0, acc01, false, false);
    acc10 = __builtin_amdgcn_wmma_f32_16x16x32_bf16(false, a1, false, b0,
                                                    (short)0, acc10, false, false);
    acc11 = __builtin_amdgcn_wmma_f32_16x16x32_bf16(false, a1, false, b1,
                                                    (short)0, acc11, false, false);
  }
  int n = lane & 15, h = lane >> 4;
#pragma unroll
  for (int r = 0; r < 8; ++r) {
    int mA = m0 + r + 8 * h;
    int mB = m0 + 16 + r + 8 * h;
    C[(long)mA * ldc + n0 + n]      = acc00[r];
    C[(long)mA * ldc + n0 + 16 + n] = acc01[r];
    C[(long)mB * ldc + n0 + n]      = acc10[r];
    C[(long)mB * ldc + n0 + 16 + n] = acc11[r];
  }
}

// ---------------------------------------------------------------------------
// LDS-staged async GEMM: C[M,N] = A[M,K] * Bt[N,K]^T (bf16 in, f32 out).
// Block = 256 threads = 8 waves, 128x128 tile; each wave computes 32x64
// (2x4 WMMA 16x16 tiles). K staged through LDS in 32-wide slices, double
// buffered, filled with GLOBAL_LOAD_ASYNC_TO_LDS_B128 (ASYNCcnt pipeline).
// LDS rows padded to 80B (40 bf16) -> conflict-free ds_load_b128.
// Requires M%128==0, N%128==0, K%32==0.
// ---------------------------------------------------------------------------
#define TILE_E 5120  // 128 rows * 40 bf16

__global__ __launch_bounds__(256) void gemm_abt_lds(
    const __bf16* __restrict__ A, const __bf16* __restrict__ Bt,
    float* __restrict__ C, int K, int lda, int ldb, int ldc) {
  __shared__ __bf16 smem[4 * TILE_E];  // [buf][A|B] : 40KB
  const int tid  = threadIdx.x;
  const int lane = tid & 31;
  const int wave = tid >> 5;
  const int wm   = wave & 3;       // M quarter (32 rows)
  const int wn   = wave >> 2;      // N half (64 cols)
  const long mblk = (long)blockIdx.y * 128;
  const long nblk = (long)blockIdx.x * 128;

  f32x8 acc[2][4] = {};

  // async fill of one K-slice (A: rows mblk..mblk+127, Bt: rows nblk..+127)
  auto fill = [&](int buf, int k0) {
    __bf16* sA = smem + (buf * 2 + 0) * TILE_E;
    __bf16* sB = smem + (buf * 2 + 1) * TILE_E;
#pragma unroll
    for (int u = 0; u < 2; ++u) {
      int q = tid + u * 256;          // chunk id 0..511
      int r = q >> 2, c = q & 3;      // row, 16B chunk within 32-elem slice
      unsigned la = (unsigned)(uintptr_t)sA + (unsigned)(r * 80 + c * 16);
      unsigned long long ga =
          (unsigned long long)(uintptr_t)(A + (mblk + r) * (long)lda + k0 + c * 8);
      asm volatile("global_load_async_to_lds_b128 %0, %1, off"
                   :: "v"(la), "v"(ga) : "memory");
      unsigned lb = (unsigned)(uintptr_t)sB + (unsigned)(r * 80 + c * 16);
      unsigned long long gb =
          (unsigned long long)(uintptr_t)(Bt + (nblk + r) * (long)ldb + k0 + c * 8);
      asm volatile("global_load_async_to_lds_b128 %0, %1, off"
                   :: "v"(lb), "v"(gb) : "memory");
    }
  };

  const int KT = K >> 5;
  fill(0, 0);
  asm volatile("s_wait_asynccnt 0x0" ::: "memory");
  __syncthreads();

  const int m = lane & 15, h = lane >> 4;
  for (int kt = 0; kt < KT; ++kt) {
    if (kt + 1 < KT) fill((kt + 1) & 1, (kt + 1) << 5);  // overlap with WMMA

    const char* sA = (const char*)(smem + ((kt & 1) * 2 + 0) * TILE_E);
    const char* sB = (const char*)(smem + ((kt & 1) * 2 + 1) * TILE_E);

    bf16x16 af[2], bfr[4];
#pragma unroll
    for (int i = 0; i < 2; ++i) {
      const char* p = sA + (wm * 32 + i * 16 + m) * 80;
      bf16x8 lo = *(const bf16x8*)(p + 16 * h);        // k = 8h..8h+7
      bf16x8 hi = *(const bf16x8*)(p + 32 + 16 * h);   // k = 16+8h..
#pragma unroll
      for (int e = 0; e < 8; ++e) { af[i][e] = lo[e]; af[i][e + 8] = hi[e]; }
    }
#pragma unroll
    for (int j = 0; j < 4; ++j) {
      const char* p = sB + (wn * 64 + j * 16 + m) * 80;
      bf16x8 lo = *(const bf16x8*)(p + 16 * h);        // k = 8h..8h+7
      bf16x8 hi = *(const bf16x8*)(p + 32 + 16 * h);   // k = 16+8h..
#pragma unroll
      for (int e = 0; e < 8; ++e) { bfr[j][e] = lo[e]; bfr[j][e + 8] = hi[e]; }
    }
#pragma unroll
    for (int i = 0; i < 2; ++i)
#pragma unroll
      for (int j = 0; j < 4; ++j)
        acc[i][j] = __builtin_amdgcn_wmma_f32_16x16x32_bf16(
            false, af[i], false, bfr[j], (short)0, acc[i][j], false, false);

    if (kt + 1 < KT) asm volatile("s_wait_asynccnt 0x0" ::: "memory");
    __syncthreads();
  }

#pragma unroll
  for (int i = 0; i < 2; ++i)
#pragma unroll
    for (int j = 0; j < 4; ++j)
#pragma unroll
      for (int r = 0; r < 8; ++r) {
        long mm = mblk + wm * 32 + i * 16 + r + 8 * h;
        long nn = nblk + wn * 64 + j * 16 + m;
        C[mm * ldc + nn] = acc[i][j][r];
      }
}

// ---------------------------------------------------------------------------
// adjacency on raw x: W[i,j] = exp(-||x_i - x_j||^2), x [NN,6]
// ---------------------------------------------------------------------------
__global__ void adj_x_kernel(const float* __restrict__ x, float* __restrict__ W) {
  int j = blockIdx.x * 256 + threadIdx.x;
  int i = blockIdx.y;
  if (j >= NN) return;
  float s = 0.f;
#pragma unroll
  for (int f = 0; f < 6; ++f) {
    float d = x[i * 6 + f] - x[j * 6 + f];
    s += d * d;
  }
  W[(long)i * NN + j] = expf(-s);
}

// adjacency epilogue on S = out*out^T: S <- exp(-(sq_i + sq_j - 2 S))
__global__ void adj_feat_kernel(float* __restrict__ S, const float* __restrict__ sq) {
  int j = blockIdx.x * 256 + threadIdx.x;
  int i = blockIdx.y;
  if (j >= NN) return;
  long idx = (long)i * NN + j;
  float v = sq[i] + sq[j] - 2.f * S[idx];
  S[idx] = expf(-v);
}

// sq[i] = sum_c X[i,c]^2 over valid channels
__global__ void rowsumsq_kernel(const float* __restrict__ X, int ld, int cv,
                                float* __restrict__ out) {
  __shared__ float red[256];
  int i = blockIdx.x, t = threadIdx.x;
  float a = 0.f;
  for (int c = t; c < cv; c += 256) {
    float v = X[(long)i * ld + c];
    a += v * v;
  }
  red[t] = a; __syncthreads();
  for (int s = 128; s > 0; s >>= 1) {
    if (t < s) red[t] += red[t + s];
    __syncthreads();
  }
  if (t == 0) out[i] = red[0];
}

// ---------------------------------------------------------------------------
// Per-row top-k sparsify (in place) + dinv = rsqrt(rowsum of kept values)
// ---------------------------------------------------------------------------
__global__ __launch_bounds__(256) void topk_kernel(float* __restrict__ W,
                                                   float* __restrict__ dinv,
                                                   int n, int kk) {
  __shared__ float row[NN];
  __shared__ float rv[256];
  __shared__ int   ri[256];
  __shared__ float selv[64];
  __shared__ int   seli[64];
  int i = blockIdx.x, t = threadIdx.x;
  for (int j = t; j < n; j += 256) row[j] = W[(long)i * n + j];
  __syncthreads();
  for (int it = 0; it < kk; ++it) {
    float bv = -1e30f; int bi = 0;
    for (int j = t; j < n; j += 256) {
      float v = row[j];
      if (v > bv) { bv = v; bi = j; }
    }
    rv[t] = bv; ri[t] = bi; __syncthreads();
    for (int s = 128; s > 0; s >>= 1) {
      if (t < s && rv[t + s] > rv[t]) { rv[t] = rv[t + s]; ri[t] = ri[t + s]; }
      __syncthreads();
    }
    if (t == 0) { selv[it] = rv[0]; seli[it] = ri[0]; row[ri[0]] = -1e30f; }
    __syncthreads();
  }
  for (int j = t; j < n; j += 256) {
    float v = 0.f;
    for (int s = 0; s < kk; ++s) if (seli[s] == j) v = selv[s];
    W[(long)i * n + j] = v;
  }
  if (t == 0) {
    float s = 0.f;
    for (int q = 0; q < kk; ++q) s += selv[q];
    dinv[i] = rsqrtf(s);
  }
}

// L[i,j] = delta_ij - W[i,j]*dinv_i*dinv_j (in place); optional bf16 mirror
__global__ void laplacian_kernel(float* __restrict__ W,
                                 const float* __restrict__ dinv,
                                 __bf16* __restrict__ outb, int n) {
  int j = blockIdx.x * 256 + threadIdx.x;
  int i = blockIdx.y;
  if (j >= n) return;
  long idx = (long)i * n + j;
  float v = ((i == j) ? 1.f : 0.f) - W[idx] * dinv[i] * dinv[j];
  W[idx] = v;
  if (outb) outb[idx] = (__bf16)v;
}

// ---------------------------------------------------------------------------
// Skinny propagation Y[i,0..F) = alpha * (L @ X)[i] - Z[i]  (F <= 8)
// ---------------------------------------------------------------------------
__global__ __launch_bounds__(256) void spmm_skinny(
    const float* __restrict__ L, const float* __restrict__ X, int ldx,
    float* __restrict__ Y, int ldy, int n, int F, float alpha,
    const float* __restrict__ Z, int ldz) {
  __shared__ float red[256 * 8];
  int i = blockIdx.x, t = threadIdx.x;
  float acc[8] = {0, 0, 0, 0, 0, 0, 0, 0};
  for (int m = t; m < n; m += 256) {
    float l = L[(long)i * n + m];
    for (int f = 0; f < F; ++f) acc[f] += l * X[(long)m * ldx + f];
  }
  for (int f = 0; f < 8; ++f) red[t * 8 + f] = acc[f];
  __syncthreads();
  for (int s = 128; s > 0; s >>= 1) {
    if (t < s)
      for (int f = 0; f < 8; ++f) red[t * 8 + f] += red[(t + s) * 8 + f];
    __syncthreads();
  }
  if (t < F) Y[(long)i * ldy + t] = alpha * red[t] - (Z ? Z[(long)i * ldz + t] : 0.f);
}

// Gram for regularizers: G[c,f] = sum_n X[n,c]*Mv[n,f]; pad f>=F with 0
__global__ __launch_bounds__(256) void gram_kernel(
    const float* __restrict__ X, int ldx, const float* __restrict__ Mv, int ldm,
    int n, int F, float* __restrict__ G) {
  __shared__ float red[256 * 8];
  int c = blockIdx.x, t = threadIdx.x;
  float acc[8] = {0, 0, 0, 0, 0, 0, 0, 0};
  for (int m = t; m < n; m += 256) {
    float xv = X[(long)m * ldx + c];
    for (int f = 0; f < F; ++f) acc[f] += xv * Mv[(long)m * ldm + f];
  }
  for (int f = 0; f < 8; ++f) red[t * 8 + f] = acc[f];
  __syncthreads();
  for (int s = 128; s > 0; s >>= 1) {
    if (t < s)
      for (int f = 0; f < 8; ++f) red[t * 8 + f] += red[(t + s) * 8 + f];
    __syncthreads();
  }
  if (t == 0)
    for (int f = 0; f < 8; ++f) G[c * 8 + f] = (f < F) ? red[f] : 0.f;
}

// single-block sum of squares over strided column; optional sqrt
__global__ __launch_bounds__(256) void sumsq_kernel(const float* __restrict__ p,
                                                    int n, int stride,
                                                    float* __restrict__ out,
                                                    int dosqrt) {
  __shared__ float red[256];
  int t = threadIdx.x;
  float a = 0.f;
  for (int i = t; i < n; i += 256) {
    float v = p[(long)i * stride];
    a += v * v;
  }
  red[t] = a; __syncthreads();
  for (int s = 128; s > 0; s >>= 1) {
    if (t < s) red[t] += red[t + s];
    __syncthreads();
  }
  if (t == 0) out[0] = dosqrt ? sqrtf(red[0]) : red[0];
}

// ---------------------------------------------------------------------------
// Epilogues / data movement
// ---------------------------------------------------------------------------
__global__ void relu_bias_kernel(float* __restrict__ Y, const float* __restrict__ b,
                                 int cv, int ld, int rows) {
  long idx = (long)blockIdx.x * 256 + threadIdx.x;
  if (idx >= (long)rows * ld) return;
  int c = (int)(idx % ld);
  Y[idx] = (c < cv) ? fmaxf(Y[idx] + b[c], 0.f) : 0.f;
}

__global__ void f32_to_bf16_kernel(const float* __restrict__ src,
                                   __bf16* __restrict__ dst, long n) {
  long idx = (long)blockIdx.x * 256 + threadIdx.x;
  if (idx < n) dst[idx] = (__bf16)src[idx];
}

// Y = alpha*P - Z (Y and Z nullable); also writes bf16 into xhat column block
__global__ void combine_kernel(const float* __restrict__ P,
                               const float* __restrict__ Z,
                               float* __restrict__ Y,
                               __bf16* __restrict__ xh, int coloff,
                               float alpha) {
  long idx = (long)blockIdx.x * 256 + threadIdx.x;
  if (idx >= (long)NN * CP) return;
  int nrow = (int)(idx / CP), c = (int)(idx % CP);
  float v = alpha * P[idx] - (Z ? Z[idx] : 0.f);
  if (Y) Y[idx] = v;
  if (xh) xh[(long)nrow * XH2K + coloff + c] = (__bf16)v;
}

// transpose + convert: src f32 [rows,cols] -> dst bf16 [cols,rows]
__global__ __launch_bounds__(256) void transpose_bf16_kernel(
    const float* __restrict__ src, __bf16* __restrict__ dst, int rows, int cols) {
  __shared__ float tile[32][33];
  int c0 = blockIdx.x * 32, r0 = blockIdx.y * 32;
  int tx = threadIdx.x, ty = threadIdx.y;
#pragma unroll
  for (int j = 0; j < 4; ++j)
    tile[ty + 8 * j][tx] = src[(long)(r0 + ty + 8 * j) * cols + c0 + tx];
  __syncthreads();
#pragma unroll
  for (int j = 0; j < 4; ++j)
    dst[(long)(c0 + ty + 8 * j) * rows + r0 + tx] = (__bf16)tile[tx][ty + 8 * j];
}

// cheb-1 operand builders
__global__ void build_xhat1_kernel(const float* __restrict__ x,
                                   const float* __restrict__ T1,
                                   const float* __restrict__ T2,
                                   __bf16* __restrict__ dst) {
  long idx = (long)blockIdx.x * 256 + threadIdx.x;
  if (idx >= (long)NN * 32) return;
  int n = (int)(idx >> 5), c = (int)(idx & 31);
  float v = 0.f;
  if (c < 6)       v = x[n * 6 + c];
  else if (c < 12) v = T1[n * 8 + (c - 6)];
  else if (c < 18) v = T2[n * 8 + (c - 12)];
  dst[idx] = (__bf16)v;
}

__global__ void build_w1t_kernel(const float* __restrict__ W1, __bf16* __restrict__ dst) {
  long idx = (long)blockIdx.x * 256 + threadIdx.x;
  if (idx >= (long)CP * 32) return;
  int o = (int)(idx >> 5), f = (int)(idx & 31);
  float v = (o < CV && f < 18) ? W1[f * CV + o] : 0.f;
  dst[idx] = (__bf16)v;
}

__global__ void build_w2t_kernel(const float* __restrict__ W2, __bf16* __restrict__ dst) {
  long idx = (long)blockIdx.x * 256 + threadIdx.x;
  if (idx >= (long)CP * XH2K) return;
  int o = (int)(idx / XH2K), kk = (int)(idx % XH2K);
  int blk = kk >> 10, c = kk & 1023;
  float v = (o < CV && c < CV) ? W2[(long)(blk * CV + c) * CV + o] : 0.f;
  dst[idx] = (__bf16)v;
}

// ---------------------------------------------------------------------------
// Reeb branch + reductions + FC
// ---------------------------------------------------------------------------
__global__ void scc_max_kernel(const float* __restrict__ out1,
                               const int* __restrict__ sccs,
                               float* __restrict__ vfr) {
  int c = blockIdx.x * 256 + threadIdx.x;
  int r = blockIdx.y;
  if (c >= CV) return;
  float m = -1e30f;
  for (int s = 0; s < S_; ++s) {
    int n = sccs[r * S_ + s];
    m = fmaxf(m, out1[(long)n * CP + c]);
  }
  vfr[r * CV + c] = m;
}

__global__ void reeb_mm_kernel(const float* __restrict__ Lr,
                               const float* __restrict__ X,
                               float* __restrict__ Y, float alpha,
                               const float* __restrict__ Z) {
  int c = blockIdx.x * 256 + threadIdx.x;
  int r = blockIdx.y;
  if (c >= CV) return;
  float s = 0.f;
  for (int r2 = 0; r2 < NR_; ++r2) s += Lr[r * NR_ + r2] * X[r2 * CV + c];
  Y[r * CV + c] = alpha * s - (Z ? Z[r * CV + c] : 0.f);
}

__global__ void reeb_cheb_out_kernel(const float* __restrict__ T0,
                                     const float* __restrict__ T1,
                                     const float* __restrict__ T2,
                                     const float* __restrict__ Wr,
                                     const float* __restrict__ br,
                                     float* __restrict__ outR) {
  int o = blockIdx.x * 256 + threadIdx.x;
  int r = blockIdx.y;
  if (o >= CV) return;
  float s = br[o];
  for (int c = 0; c < CV; ++c) s += T0[r * CV + c] * Wr[(long)c * CV + o];
  for (int c = 0; c < CV; ++c) s += T1[r * CV + c] * Wr[(long)(CV + c) * CV + o];
  for (int c = 0; c < CV; ++c) s += T2[r * CV + c] * Wr[(long)(2 * CV + c) * CV + o];
  outR[r * CV + o] = fmaxf(s, 0.f);
}

__global__ void maxcols_kernel(const float* __restrict__ X, int rows, int ld,
                               int ncols, float* __restrict__ dst) {
  int c = blockIdx.x * 256 + threadIdx.x;
  if (c >= ncols) return;
  float m = -1e30f;
  for (int r = 0; r < rows; ++r) m = fmaxf(m, X[(long)r * ld + c]);
  dst[c] = m;
}

__global__ void fc_kernel(const float* __restrict__ in, const float* __restrict__ W,
                          const float* __restrict__ b, float* __restrict__ out,
                          int nin, int nout, int relu) {
  int j = blockIdx.x * 256 + threadIdx.x;
  if (j >= nout) return;
  float s = b[j];
  for (int f = 0; f < nin; ++f) s += in[f] * W[(long)f * nout + j];
  out[j] = relu ? fmaxf(s, 0.f) : s;
}

// ---------------------------------------------------------------------------
// Orchestration
// ---------------------------------------------------------------------------
extern "C" void kernel_launch(void* const* d_in, const int* in_sizes, int n_in,
                              void* d_out, int out_size, void* d_ws, size_t ws_size,
                              hipStream_t stream) {
  (void)in_sizes; (void)n_in; (void)out_size; (void)ws_size;
  const float* x     = (const float*)d_in[0];
  const float* Lr    = (const float*)d_in[5];
  const int*   sccs  = (const int*)d_in[6];
  const float* W1    = (const float*)d_in[7];
  const float* b1    = (const float*)d_in[8];
  const float* W2    = (const float*)d_in[9];
  const float* b2    = (const float*)d_in[10];
  const float* Wr    = (const float*)d_in[11];
  const float* br    = (const float*)d_in[12];
  const float* fc1w  = (const float*)d_in[13];
  const float* fc1b  = (const float*)d_in[14];
  const float* fc2w  = (const float*)d_in[15];
  const float* fc2b  = (const float*)d_in[16];
  const float* fc3w  = (const float*)d_in[17];
  const float* fc3b  = (const float*)d_in[18];
  float* out = (float*)d_out;

  // workspace carving (256B aligned)
  char* w = (char*)d_ws; size_t off = 0;
  auto alloc = [&](size_t bytes) -> void* {
    void* p = w + off; off += (bytes + 255) & ~(size_t)255; return p;
  };
  float*  bufA   = (float*) alloc((size_t)NN * NN * 4);     // adj/L (64MB)
  __bf16* L2b    = (__bf16*)alloc((size_t)NN * NN * 2);     // L2 bf16 (32MB)
  float*  outF   = (float*) alloc((size_t)NN * CP * 4);     // out1 padded
  __bf16* outB   = (__bf16*)alloc((size_t)NN * CP * 2);
  float*  TA     = (float*) alloc((size_t)NN * CP * 4);
  float*  TB     = (float*) alloc((size_t)NN * CP * 4);
  float*  Pb     = (float*) alloc((size_t)NN * CP * 4);
  __bf16* TtB    = (__bf16*)alloc((size_t)CP * NN * 2);
  __bf16* xhatB  = (__bf16*)alloc((size_t)NN * XH2K * 2);   // 48MB
  float*  out2F  = (float*) alloc((size_t)NN * CP * 4);
  __bf16* W2tB   = (__bf16*)alloc((size_t)CP * XH2K * 2);
  __bf16* xhat1B = (__bf16*)alloc((size_t)NN * 32 * 2);
  __bf16* W1tB   = (__bf16*)alloc((size_t)CP * 32 * 2);
  float*  dinv   = (float*) alloc(NN * 4);
  float*  sq2    = (float*) alloc(NN * 4);
  float*  Tx1    = (float*) alloc((size_t)NN * 8 * 4);
  float*  Tx2    = (float*) alloc((size_t)NN * 8 * 4);
  float*  M2     = (float*) alloc((size_t)NN * 8 * 4);
  float*  G      = (float*) alloc((size_t)CV * 8 * 4);
  float*  vfr    = (float*) alloc((size_t)NR_ * CV * 4);
  float*  TxR1   = (float*) alloc((size_t)NR_ * CV * 4);
  float*  TxR2   = (float*) alloc((size_t)NR_ * CV * 4);
  float*  outR   = (float*) alloc((size_t)NR_ * CV * 4);
  float*  ofv    = (float*) alloc(2048 * 4);
  float*  h1     = (float*) alloc(512 * 4);
  float*  h2     = (float*) alloc(128 * 4);

  dim3 b256(256);
  dim3 gRow(NN / 256, NN);              // (16, 4096)
  dim3 gCols4(4, NR_);                  // (ceil(1000/256), 20)

  // --- stage 1: graph on x, L1, cheb1 ---------------------------------------
  adj_x_kernel<<<gRow, b256, 0, stream>>>(x, bufA);
  topk_kernel<<<NN, 256, 0, stream>>>(bufA, dinv, NN, 40);
  laplacian_kernel<<<gRow, b256, 0, stream>>>(bufA, dinv, (__bf16*)nullptr, NN);
  spmm_skinny<<<NN, 256, 0, stream>>>(bufA, x, 6, Tx1, 8, NN, 6, 1.f, nullptr, 0);
  spmm_skinny<<<NN, 256, 0, stream>>>(bufA, Tx1, 8, Tx2, 8, NN, 6, 2.f, x, 6);
  build_xhat1_kernel<<<(NN * 32) / 256, b256, 0, stream>>>(x, Tx1, Tx2, xhat1B);
  build_w1t_kernel<<<(CP * 32) / 256, b256, 0, stream>>>(W1, W1tB);
  gemm_abt_lds<<<dim3(CP / 128, NN / 128), b256, 0, stream>>>(
      xhat1B, W1tB, outF, 32, 32, 32, CP);
  relu_bias_kernel<<<(NN * CP) / 256, b256, 0, stream>>>(outF, b1, CV, CP, NN);

  // reg0 = ||out1^T (L1 x)||_F   (L1 x == Tx1)
  gram_kernel<<<CV, 256, 0, stream>>>(outF, CP, Tx1, 8, NN, 6, G);
  sumsq_kernel<<<1, 256, 0, stream>>>(G, CV * 8, 1, out + 40, 1);

  // Reeb pooling input (uses out1)
  scc_max_kernel<<<gCols4, b256, 0, stream>>>(outF, sccs, vfr);

  // --- stage 2: graph on out1 features (the big GEMM), L2 -------------------
  f32_to_bf16_kernel<<<(NN * CP) / 256, b256, 0, stream>>>(outF, outB, (long)NN * CP);
  gemm_abt_lds<<<dim3(NN / 128, NN / 128), b256, 0, stream>>>(
      outB, outB, bufA, CP, CP, CP, NN);                          // S = out*out^T
  rowsumsq_kernel<<<NN, 256, 0, stream>>>(outF, CP, CV, sq2);
  adj_feat_kernel<<<gRow, b256, 0, stream>>>(bufA, sq2);
  topk_kernel<<<NN, 256, 0, stream>>>(bufA, dinv, NN, 40);
  laplacian_kernel<<<gRow, b256, 0, stream>>>(bufA, dinv, L2b, NN);
  spmm_skinny<<<NN, 256, 0, stream>>>(bufA, x, 6, M2, 8, NN, 6, 1.f, nullptr, 0); // L2 x

  // --- cheb-2: T0=out1; T1=L2 T0; Tk=2 L2 T_{k-1} - T_{k-2}; WMMA props -----
  dim3 gT(CP / 32, NN / 32), bT(32, 8);
  dim3 gProp(CP / 128, NN / 128);
  long ncomb = (long)NN * CP / 256;
  // block 0: out1
  combine_kernel<<<ncomb, b256, 0, stream>>>(outF, nullptr, nullptr, xhatB, 0, 1.f);
  // k=1
  transpose_bf16_kernel<<<gT, bT, 0, stream>>>(outF, TtB, NN, CP);
  gemm_abt_lds<<<gProp, b256, 0, stream>>>(L2b, TtB, Pb, NN, NN, NN, CP);
  combine_kernel<<<ncomb, b256, 0, stream>>>(Pb, nullptr, TA, xhatB, 1 * CP, 1.f);
  // k=2
  transpose_bf16_kernel<<<gT, bT, 0, stream>>>(TA, TtB, NN, CP);
  gemm_abt_lds<<<gProp, b256, 0, stream>>>(L2b, TtB, Pb, NN, NN, NN, CP);
  combine_kernel<<<ncomb, b256, 0, stream>>>(Pb, outF, TB, xhatB, 2 * CP, 2.f);
  // k=3 (TA <- 2P - TA)
  transpose_bf16_kernel<<<gT, bT, 0, stream>>>(TB, TtB, NN, CP);
  gemm_abt_lds<<<gProp, b256, 0, stream>>>(L2b, TtB, Pb, NN, NN, NN, CP);
  combine_kernel<<<ncomb, b256, 0, stream>>>(Pb, TA, TA, xhatB, 3 * CP, 2.f);
  // k=4 (TB <- 2P - TB)
  transpose_bf16_kernel<<<gT, bT, 0, stream>>>(TA, TtB, NN, CP);
  gemm_abt_lds<<<gProp, b256, 0, stream>>>(L2b, TtB, Pb, NN, NN, NN, CP);
  combine_kernel<<<ncomb, b256, 0, stream>>>(Pb, TB, TB, xhatB, 4 * CP, 2.f);
  // k=5 (TA <- 2P - TA)
  transpose_bf16_kernel<<<gT, bT, 0, stream>>>(TB, TtB, NN, CP);
  gemm_abt_lds<<<gProp, b256, 0, stream>>>(L2b, TtB, Pb, NN, NN, NN, CP);
  combine_kernel<<<ncomb, b256, 0, stream>>>(Pb, TA, TA, xhatB, 5 * CP, 2.f);

  // out2 = relu(xhat @ W2 + b2)
  build_w2t_kernel<<<((long)CP * XH2K) / 256, b256, 0, stream>>>(W2, W2tB);
  gemm_abt_lds<<<dim3(CP / 128, NN / 128), b256, 0, stream>>>(
      xhatB, W2tB, out2F, XH2K, XH2K, XH2K, CP);
  relu_bias_kernel<<<(NN * CP) / 256, b256, 0, stream>>>(out2F, b2, CV, CP, NN);

  // reg1 = ||out2^T (L2 x)||_F
  gram_kernel<<<CV, 256, 0, stream>>>(out2F, CP, M2, 8, NN, 6, G);
  sumsq_kernel<<<1, 256, 0, stream>>>(G, CV * 8, 1, out + 41, 1);

  // global max pools
  maxcols_kernel<<<4, b256, 0, stream>>>(out2F, NN, CP, CV, ofv + CV);

  // --- Reeb branch ----------------------------------------------------------
  reeb_mm_kernel<<<gCols4, b256, 0, stream>>>(Lr, vfr, TxR1, 1.f, nullptr);
  reeb_mm_kernel<<<gCols4, b256, 0, stream>>>(Lr, TxR1, TxR2, 2.f, vfr);
  reeb_cheb_out_kernel<<<gCols4, b256, 0, stream>>>(vfr, TxR1, TxR2, Wr, br, outR);
  maxcols_kernel<<<4, b256, 0, stream>>>(outR, NR_, CV, CV, ofv);

  // --- FC head --------------------------------------------------------------
  fc_kernel<<<2, b256, 0, stream>>>(ofv, fc1w, fc1b, h1, 2 * CV, 512, 1);
  fc_kernel<<<1, b256, 0, stream>>>(h1, fc2w, fc2b, h2, 512, 128, 1);
  fc_kernel<<<1, b256, 0, stream>>>(h2, fc3w, fc3b, out, 128, 40, 0);

  // --- remaining regularizers ----------------------------------------------
  sumsq_kernel<<<1, 256, 0, stream>>>(fc1w, 2 * CV, 512, out + 42, 0);
  sumsq_kernel<<<1, 256, 0, stream>>>(fc1b, 1, 1, out + 43, 0);
  sumsq_kernel<<<1, 256, 0, stream>>>(fc2w, 512, 128, out + 44, 0);
  sumsq_kernel<<<1, 256, 0, stream>>>(fc2b, 1, 1, out + 45, 0);
  sumsq_kernel<<<1, 256, 0, stream>>>(fc3w, 128, 40, out + 46, 0);
  sumsq_kernel<<<1, 256, 0, stream>>>(fc3b, 1, 1, out + 47, 0);
}
